// Self_Attn_1520418422868
// MI455X (gfx1250) — matrix operand order repaired and checked by reading, not compiled, not run on hardware
//
#include <hip/hip_runtime.h>

// ---------------------------------------------------------------------------
// SAGAN self-attention, fused flash-style for gfx1250 (wave32, WMMA f16).
//   B=16, C=64, H=W=64, N=4096, M=1024
// Workspace layout (3.5 MB):
//   theta_t : f16 [B][4096][16]  (k 0..7 | 8 zero halves -> K-padded operand
//                                 loaded with a per-half address offset)
//   phi_t   : f16 [B][1024][16]  (same record shape)
//   g_ws    : f16 [B][32][1024]  (m-contiguous -> LDS A-operand ds_load_b64)
//
// Shuffle-free repack: WMMA D-tile row order == A-operand row order, so the
// phi/g A-operands gather a permuted row set (rows 8..15 -> +8) such that
// v_cvt_pk_rtz of D-register pairs lands directly in the next GEMM's f16
// B-operand in the contiguous k order used here (Vj: low lanes K=2j,2j+1; high lanes K=16+2j,17+2j).
// ---------------------------------------------------------------------------

typedef __attribute__((ext_vector_type(8)))  float    v8f;
typedef __attribute__((ext_vector_type(16))) _Float16 v16h;

union V16H { v16h v; unsigned u[8]; };

__device__ __forceinline__ unsigned pkh(float a, float b) {
  return __builtin_bit_cast(unsigned, __builtin_amdgcn_cvt_pkrtz(a, b));
}

// ---------------------------------------------------------------------------
// Kernel A: 1x1-conv projections + 2x2 maxpool. One block = one batch b and
// one pooled row (2 input rows = 128 spatial positions, 128 threads).
// ---------------------------------------------------------------------------
__global__ __launch_bounds__(128) void proj_pool_kernel(
    const float* __restrict__ x,
    const float* __restrict__ Wt, const float* __restrict__ Wp,
    const float* __restrict__ Wg,
    _Float16* __restrict__ theta_t,
    _Float16* __restrict__ phi_t,
    _Float16* __restrict__ g_ws)
{
  __shared__ __align__(16) float Wall[64][48];   // [c][theta8 | phi8 | g32]
  __shared__ float pool[128][40];
  const int b   = blockIdx.x >> 5;
  const int rp  = blockIdx.x & 31;
  const int tid = threadIdx.x;

  for (int e = tid; e < 512;  e += 128) Wall[e & 63][      (e >> 6)] = Wt[e];
  for (int e = tid; e < 512;  e += 128) Wall[e & 63][8  +  (e >> 6)] = Wp[e];
  for (int e = tid; e < 2048; e += 128) Wall[e & 63][16 +  (e >> 6)] = Wg[e];
  __syncthreads();

  const int n  = (rp*2 + (tid >> 6))*64 + (tid & 63);
  const float* xb = x + (long)b*64*4096 + n;

  float acc[48];
#pragma unroll
  for (int j = 0; j < 48; ++j) acc[j] = 0.f;

  for (int c = 0; c < 64; ++c) {
    float xv = xb[(long)c*4096];                 // coalesced across block
    const float4* w4 = (const float4*)&Wall[c][0];   // ds_load_b128 x12
#pragma unroll
    for (int q = 0; q < 12; ++q) {
      float4 w = w4[q];
      acc[4*q+0] = fmaf(w.x, xv, acc[4*q+0]);
      acc[4*q+1] = fmaf(w.y, xv, acc[4*q+1]);
      acc[4*q+2] = fmaf(w.z, xv, acc[4*q+2]);
      acc[4*q+3] = fmaf(w.w, xv, acc[4*q+3]);
    }
  }

  // theta record: 8 packed f16 + 8 zero halves (K padding lives in memory)
  const uint4 z4 = make_uint4(0u, 0u, 0u, 0u);
  uint4 tv;
  tv.x = pkh(acc[0], acc[1]); tv.y = pkh(acc[2], acc[3]);
  tv.z = pkh(acc[4], acc[5]); tv.w = pkh(acc[6], acc[7]);
  *(uint4*)(theta_t + ((long)b*4096 + n)*16)     = tv;
  *(uint4*)(theta_t + ((long)b*4096 + n)*16 + 8) = z4;

#pragma unroll
  for (int j = 0; j < 40; ++j) pool[tid][j] = acc[8 + j];
  __syncthreads();

  // 2x2 maxpool: 32 pooled cols x 40 channels, 10 per thread
  const int cp = tid & 31, grp = tid >> 5;
  const int m  = rp*32 + cp;
#pragma unroll
  for (int jj = 0; jj < 10; ++jj) {
    int j = grp*10 + jj;
    float v = fmaxf(fmaxf(pool[2*cp][j],    pool[2*cp+1][j]),
                    fmaxf(pool[64+2*cp][j], pool[65+2*cp][j]));
    if (j < 8) phi_t[((long)b*1024 + m)*16 + j]       = (_Float16)v;
    else       g_ws [((long)b*32 + (j-8))*1024 + m]   = (_Float16)v;
  }
  if (grp == 0)  // zero padding halves of the phi record
    *(uint4*)(phi_t + ((long)b*1024 + m)*16 + 8) = z4;
}

// ---------------------------------------------------------------------------
// Kernel B: fused attention. Block = (batch b, 128 n-rows); 8 waves, each
// wave owns 16 n-columns. S^T orientation => softmax axis is in-lane.
// ---------------------------------------------------------------------------
__global__ __launch_bounds__(256) void flash_attn_kernel(
    const float* __restrict__ x,
    const _Float16* __restrict__ theta_t,
    const _Float16* __restrict__ phi_t,
    const _Float16* __restrict__ g_ws,
    const float* __restrict__ Wattn,
    const float* __restrict__ sigma,
    float* __restrict__ out)
{
  __shared__ _Float16 gl[32*1024];               // g[b] staged: 64 KB
  const int b   = blockIdx.x >> 5;
  const int tid = threadIdx.x;
  {
    const uint4* src = (const uint4*)(g_ws + (long)b*32*1024);
    uint4* dst = (uint4*)gl;
    for (int i = tid; i < 4096; i += 256) dst[i] = src[i];
  }
  __syncthreads();

  const int  lane = tid & 31;
  const int  ln   = lane & 15;
  const bool lop  = lane < 16;
  const int  hof  = lop ? 0 : 8;                 // high lanes read the zeros
  const int  nb   = (blockIdx.x & 31)*128 + (tid >> 5)*16;
  // permuted row id: logical rows 8..15 carry rows +8 of the next GEMM's K
  const int  rperm = (ln < 8) ? ln : (ln + 8);

  // theta B-operand (K=8, zero-padded via record layout); fixed for the loop
  V16H Bt;
  {
    uint4 u = *(const uint4*)(theta_t + ((long)b*4096 + nb + ln)*16 + hof);
    Bt.u[0] = u.x; Bt.u[1] = u.y; Bt.u[2] = u.z; Bt.u[3] = u.w;
    Bt.u[4] = 0u; Bt.u[5] = 0u; Bt.u[6] = 0u; Bt.u[7] = 0u;
  }
  const _Float16* phb = phi_t + (long)b*1024*16;

  v8f acc0 = {}; v8f acc1 = {};                  // attn_g tiles (permuted rows)
  float rs = 0.f;                                // per-column running sum(exp)

  for (int ch = 0; ch < 32; ++ch) {
    const int mb = ch*32;
    // phi A-operands; subtile0 rows -> m {0..7,16..23}, subtile1 -> {8..15,24..31}
    uint4 a0 = *(const uint4*)(phb + (long)(mb + rperm    )*16 + hof);
    uint4 a1 = *(const uint4*)(phb + (long)(mb + rperm + 8)*16 + hof);
    V16H A0, A1;
    A0.u[0]=a0.x; A0.u[1]=a0.y; A0.u[2]=a0.z; A0.u[3]=a0.w;
    A0.u[4]=0u; A0.u[5]=0u; A0.u[6]=0u; A0.u[7]=0u;
    A1.u[0]=a1.x; A1.u[1]=a1.y; A1.u[2]=a1.z; A1.u[3]=a1.w;
    A1.u[4]=0u; A1.u[5]=0u; A1.u[6]=0u; A1.u[7]=0u;

    v8f c0 = {}; v8f c1 = {};                    // S^T tiles, rows = permuted m
    c0 = __builtin_amdgcn_wmma_f32_16x16x32_f16(false, A0.v, false, Bt.v, (short)0, c0, false, false);
    c1 = __builtin_amdgcn_wmma_f32_16x16x32_f16(false, A1.v, false, Bt.v, (short)0, c1, false, false);

    // max-free softmax numerator (logits bounded; shift-invariant => exact)
#pragma unroll
    for (int r = 0; r < 8; ++r) {
      c0[r] = __expf(c0[r]); c1[r] = __expf(c1[r]);
      rs += c0[r] + c1[r];
    }

    // D-tile pairs pack DIRECTLY into the f16 B-operand layout (no shuffles)
    V16H P;
#pragma unroll
    for (int j = 0; j < 4; ++j) {
      P.u[j]     = pkh(c0[2*j], c0[2*j+1]);      // K=2j,2j+1 | 16+2j,17+2j
      P.u[j + 4] = pkh(c1[2*j], c1[2*j+1]);      // K=8+2j    | 24+2j
    }

    // g A-operands from LDS, rows gathered with the same permutation
    V16H G0, G1;
#pragma unroll
    for (int p = 0; p < 4; ++p) {
      const int moff = 4*p + (lop ? 0 : 16);
      uint2 ga = *(const uint2*)(gl + (rperm    )*1024 + mb + moff);
      uint2 gb = *(const uint2*)(gl + (rperm + 8)*1024 + mb + moff);
      G0.u[2*p] = ga.x; G0.u[2*p+1] = ga.y;
      G1.u[2*p] = gb.x; G1.u[2*p+1] = gb.y;
    }
    acc0 = __builtin_amdgcn_wmma_f32_16x16x32_f16(false, G0.v, false, P.v, (short)0, acc0, false, false);
    acc1 = __builtin_amdgcn_wmma_f32_16x16x32_f16(false, G1.v, false, P.v, (short)0, acc1, false, false);
  }

  // normalize (1/sum) and fold in sigma; per-column scalar lives per-lane
  float rtot = rs + __shfl_xor(rs, 16, 32);
  float rinv = sigma[0] / rtot;
#pragma unroll
  for (int r = 0; r < 8; ++r) { acc0[r] *= rinv; acc1[r] *= rinv; }

  // acc rows are permuted gch => packs land directly in B-operand layout
  V16H B3;
#pragma unroll
  for (int j = 0; j < 4; ++j) {
    B3.u[j]     = pkh(acc0[2*j], acc0[2*j+1]);
    B3.u[j + 4] = pkh(acc1[2*j], acc1[2*j+1]);
  }

  // final projection W_attn[64x32] @ attn_g + residual
#pragma unroll
  for (int t = 0; t < 4; ++t) {
    V16H A3;
#pragma unroll
    for (int p = 0; p < 4; ++p) {
      const int goff = 4*p + (lop ? 0 : 16);
      float4 w = *(const float4*)(Wattn + (t*16 + ln)*32 + goff);
      A3.u[2*p]   = pkh(w.x, w.y);
      A3.u[2*p+1] = pkh(w.z, w.w);
    }
    v8f d = {};
    d = __builtin_amdgcn_wmma_f32_16x16x32_f16(false, A3.v, false, B3.v, (short)0, d, false, false);
    const int cb = t*16 + (lop ? 0 : 8);
#pragma unroll
    for (int r = 0; r < 8; ++r) {
      long idx = ((long)(b*64 + cb + r))*4096 + nb + ln;
      out[idx] = x[idx] + d[r];
    }
  }
}

// ---------------------------------------------------------------------------
extern "C" void kernel_launch(void* const* d_in, const int* in_sizes, int n_in,
                              void* d_out, int out_size, void* d_ws, size_t ws_size,
                              hipStream_t stream) {
  const float* x   = (const float*)d_in[0];
  const float* Wt  = (const float*)d_in[1];
  const float* Wp  = (const float*)d_in[2];
  const float* Wg  = (const float*)d_in[3];
  const float* Wa  = (const float*)d_in[4];
  const float* sig = (const float*)d_in[5];
  float* out = (float*)d_out;

  char* ws = (char*)d_ws;                        // needs 3.5 MB
  _Float16* theta_t = (_Float16*)(ws);                           // 2 MB
  _Float16* phi_t   = (_Float16*)(ws + (size_t)2048*1024);       // 512 KB
  _Float16* g_ws    = (_Float16*)(ws + (size_t)2560*1024);       // 1 MB

  proj_pool_kernel<<<512, 128, 0, stream>>>(x, Wt, Wp, Wg, theta_t, phi_t, g_ws);
  flash_attn_kernel<<<512, 256, 0, stream>>>(x, theta_t, phi_t, g_ws, Wa, sig, out);
}
